// QLSTM_65481071403711
// MI455X (gfx1250) — matrix operand order repaired and checked by reading, not compiled, not run on hardware
//
#include <hip/hip_runtime.h>
#include <hip/hip_bf16.h>

// ---------------------------------------------------------------------------
// QLSTM for MI455X (gfx1250): bf16 WMMA everywhere, fp32 accumulate.
//  Phase 1 (parallel, WMMA):
//      [xproj | xgate] = x[t,b,:] @ [W1[:256,:] | W{f,i,g,o}[:256] | 0] + bias
//      (144-column extended B matrix; one GEMM covers both projections)
//  Phase 2 (sequential, persistent single workgroup, 32 waves, all state in
//           one WGP's 320 KB LDS):
//      [h1 | hgate] = hx @ [W1h | Whg | 0]   (WMMA)
//      q = h1 @ W2 + b2                      (WMMA)
//      qgate = mean cumprod cos ; mix ; cx/hx update
// B matrices are pre-swizzled into WMMA fragment layout in LDS so every
// fragment load is one 32-byte contiguous LDS read (2x ds_load_b128); the A
// layout's one contiguous 16-element K-run per lane coalesces the same way.
// ---------------------------------------------------------------------------

#define SEQ     512
#define BATCH   256
#define IN_DIM  256
#define HID_DIM 256
#define MLP_HID 128
#define NQ      8
#define RTOT    (SEQ * BATCH)   // 131072 rows
#define NT1     9               // 8 main col-tiles + 1 gate col-tile (cols 128..143)

typedef __attribute__((ext_vector_type(16))) __bf16 v16bf;
typedef __attribute__((ext_vector_type(8)))  float  v8f;

// ---- A-fragment loader (16x32 bf16, linear row-major source) --------------
// Lane layout (this code's k order, not the ISA table's): row = lane&15, half = lane>>4; per lane the 16
// elements are one contiguous 16-element K-run (k = 16*half + e) -> 2x ds_load_b128.
__device__ __forceinline__ v16bf load_a_tile(const __bf16* base, int stride, int lane) {
    int row = lane & 15;
    int hv  = lane >> 4;
    v16bf a;
#pragma unroll
    for (int p = 0; p < 8; ++p) {
        int k0 = 2 * p + 16 * hv;
        a[2 * p]     = base[row * stride + k0];
        a[2 * p + 1] = base[row * stride + k0 + 1];
    }
    return a;
}

// ---- B-fragment loader: fragment-layout LDS, 32 contiguous bytes per lane --
__device__ __forceinline__ v16bf load_b_frag(const __bf16* frag, int lane) {
    return *(const v16bf*)(frag + lane * 16);
}

__device__ __forceinline__ float sigmoidf_(float x) { return 1.0f / (1.0f + __expf(-x)); }

// ---------------------------------------------------------------------------
// Kernel 1: [xproj | xg] = x @ [W1x | Wgx | 0] + bias   (R x 144), bf16 WMMA.
// 256 threads = 8 waves per block; block handles 16 rows; wave -> col tiles
// {wave, wave+8}. The extended B (144 cols) is staged pre-swizzled in LDS.
// ---------------------------------------------------------------------------
__global__ void qlstm_xproj_gemm(const float* __restrict__ x,
                                 const float* __restrict__ W1,
                                 const float* __restrict__ b1,
                                 const float* __restrict__ Wf, const float* __restrict__ bfv,
                                 const float* __restrict__ Wi, const float* __restrict__ biv,
                                 const float* __restrict__ Wg, const float* __restrict__ bgv,
                                 const float* __restrict__ Wo, const float* __restrict__ bov,
                                 float* __restrict__ xproj,   // (R,128)
                                 float* __restrict__ xg) {    // (R,4)
    extern __shared__ char smem[];
    __bf16* sWF = (__bf16*)smem;                 // 8*9 tiles * 512 = 36864 bf16
    __bf16* sX  = sWF + 8 * NT1 * 512;           // 16 x 256 bf16

    int tid = threadIdx.x;
    // Pre-swizzle extended B into fragment layout:
    // frag[(kt*NT1+nt)*512 + l*16 + e] = B[kt*32 + 16*(l>>4) + e][nt*16 + (l&15)]
    for (int i = tid; i < 8 * NT1 * 512; i += 256) {
        int tileIdx = i >> 9;
        int rem = i & 511;
        int l = rem >> 4, e = rem & 15;
        int kt = tileIdx / NT1, nt = tileIdx % NT1;
        int k = kt * 32 + 16 * (l >> 4) + e;     // 0..255 (x dim)
        int n = nt * 16 + (l & 15);              // 0..143
        float v;
        if      (n < 128)  v = W1[k * MLP_HID + n];
        else if (n == 128) v = Wf[k];
        else if (n == 129) v = Wi[k];
        else if (n == 130) v = Wg[k];
        else if (n == 131) v = Wo[k];
        else               v = 0.0f;
        sWF[i] = (__bf16)v;
    }
    long r0 = (long)blockIdx.x * 16;
    for (int i = tid; i < 16 * IN_DIM; i += 256) {
        int rr = i >> 8, kk = i & 255;
        sX[rr * IN_DIM + kk] = (__bf16)x[(r0 + rr) * IN_DIM + kk];
    }
    __syncthreads();

    int wave = tid >> 5, lane = tid & 31;
    for (int nt = wave; nt < NT1; nt += 8) {
        v8f c = {};
#pragma unroll
        for (int kt = 0; kt < 8; ++kt) {
            v16bf a = load_a_tile(sX + kt * 32, IN_DIM, lane);
            v16bf b = load_b_frag(sWF + (kt * NT1 + nt) * 512, lane);
            c = __builtin_amdgcn_wmma_f32_16x16x32_bf16(false, a, false, b,
                                                        (short)0, c, false, false);
        }
        if (nt < 8) {
            int n = nt * 16 + (lane & 15);
            float bias = b1[n];
#pragma unroll
            for (int r = 0; r < 8; ++r) {
                int m = r + 8 * (lane >> 4);
                xproj[(r0 + m) * MLP_HID + n] = c[r] + bias;
            }
        } else if ((lane & 15) < 4) {                        // gate columns 128..131
            int g = lane & 15;
            float gbias = (g == 0) ? bfv[0] : (g == 1) ? biv[0]
                         : (g == 2) ? bgv[0] : bov[0];
#pragma unroll
            for (int r = 0; r < 8; ++r) {
                int m = r + 8 * (lane >> 4);
                xg[(r0 + m) * 4 + g] = c[r] + gbias;
            }
        }
    }
}

// ---------------------------------------------------------------------------
// Kernel 2: persistent recurrent kernel, one workgroup of 1024 threads
// (32 waves on one WGP). LDS layout (319,616 B of 320 KB):
// ---------------------------------------------------------------------------
#define LDS_HX_OFF   0                                        // 131072 B (256x256 bf16, linear)
#define LDS_W1HF_OFF (LDS_HX_OFF   + HID_DIM * HID_DIM * 2)   //  73728 B (8*9 B-frag tiles)
#define LDS_H1_OFF   (LDS_W1HF_OFF + 8 * NT1 * 512 * 2)       //  65536 B (256x128 bf16, linear)
#define LDS_W2F_OFF  (LDS_H1_OFF   + BATCH * MLP_HID * 2)     //   8192 B (4*2 B-frag tiles)
#define LDS_Q_OFF    (LDS_W2F_OFF  + 8 * 512 * 2)             //  32768 B (256x32 f32)
#define LDS_GPRE_OFF (LDS_Q_OFF    + BATCH * 32 * 4)          //   4096 B (256x4 f32 hgate)
#define LDS_GATE_OFF (LDS_GPRE_OFF + BATCH * 4 * 4)           //   4096 B (4x256 f32 mixed)
#define LDS_B2_OFF   (LDS_GATE_OFF + 4 * BATCH * 4)           //    128 B
#define LDS_TOTAL    (LDS_B2_OFF   + 32 * 4)

__global__ void __launch_bounds__(1024, 1)
qlstm_recurrent(const float* __restrict__ xproj,   // (R,128)
                const float* __restrict__ xg,      // (R,4) incl. biases
                const float* __restrict__ W1,      // (512,128)
                const float* __restrict__ W2,      // (128,32)
                const float* __restrict__ b2,      // (32)
                const float* __restrict__ Wf, const float* __restrict__ Wi,
                const float* __restrict__ Wg, const float* __restrict__ Wo,
                float* __restrict__ cxws,          // (256,256) scratch
                float* __restrict__ out) {         // outputs | hx | cx
    extern __shared__ char smem[];
    __bf16* sHx   = (__bf16*)(smem + LDS_HX_OFF);
    __bf16* sW1hF = (__bf16*)(smem + LDS_W1HF_OFF);
    __bf16* sH1   = (__bf16*)(smem + LDS_H1_OFF);
    __bf16* sW2F  = (__bf16*)(smem + LDS_W2F_OFF);
    float*  sQ    = (float*) (smem + LDS_Q_OFF);
    float*  sGpre = (float*) (smem + LDS_GPRE_OFF);
    float*  sGate = (float*) (smem + LDS_GATE_OFF);
    float*  sB2   = (float*) (smem + LDS_B2_OFF);

    int tid  = threadIdx.x;
    int wave = tid >> 5;
    int lane = tid & 31;

    // ---- preload: swizzle [W1h | Whg | 0] (256 x 144) to B-frag layout ----
    for (int i = tid; i < 8 * NT1 * 512; i += 1024) {
        int tileIdx = i >> 9;
        int rem = i & 511;
        int l = rem >> 4, e = rem & 15;
        int kt = tileIdx / NT1, nt = tileIdx % NT1;
        int k = kt * 32 + 16 * (l >> 4) + e;     // 0..255 (hx dim)
        int n = nt * 16 + (l & 15);              // 0..143
        float v;
        if      (n < 128)  v = W1[(IN_DIM + k) * MLP_HID + n];
        else if (n == 128) v = Wf[IN_DIM + k];
        else if (n == 129) v = Wi[IN_DIM + k];
        else if (n == 130) v = Wg[IN_DIM + k];
        else if (n == 131) v = Wo[IN_DIM + k];
        else               v = 0.0f;
        sW1hF[i] = (__bf16)v;
    }
    // ---- preload: swizzle W2 (128 x 32) to B-frag layout ----
    for (int i = tid; i < 8 * 512; i += 1024) {
        int tileIdx = i >> 9;
        int rem = i & 511;
        int l = rem >> 4, e = rem & 15;
        int kt = tileIdx >> 1, nt = tileIdx & 1;
        int k = kt * 32 + 16 * (l >> 4) + e;     // 0..127
        int n = nt * 16 + (l & 15);              // 0..31
        sW2F[i] = (__bf16)W2[k * 32 + n];
    }
    if (tid < 32) sB2[tid] = b2[tid];
    for (int i = tid; i < BATCH * HID_DIM; i += 1024) {
        sHx[i]  = (__bf16)0.0f;
        cxws[i] = 0.0f;
    }
    __syncthreads();

    for (int t = 0; t < SEQ; ++t) {
        const float* xproj_t = xproj + (long)t * BATCH * MLP_HID;
        const float* xg_t    = xg    + (long)t * BATCH * 4;

        // ---- Stage A: [h1 | hgate] = hx @ [W1h | Whg]  (WMMA from LDS) ----
        // 256 x 144 output = 16 row-tiles x 9 col-tiles = 144 tiles.
        for (int tile = wave; tile < 16 * NT1; tile += 32) {
            int mt = tile / NT1, nt = tile % NT1;
            v8f c = {};
#pragma unroll
            for (int kt = 0; kt < 8; ++kt) {
                v16bf a = load_a_tile(sHx + mt * 16 * HID_DIM + kt * 32, HID_DIM, lane);
                v16bf b = load_b_frag(sW1hF + (kt * NT1 + nt) * 512, lane);
                c = __builtin_amdgcn_wmma_f32_16x16x32_bf16(false, a, false, b,
                                                            (short)0, c, false, false);
            }
            if (nt < 8) {
                int n = nt * 16 + (lane & 15);
#pragma unroll
                for (int r = 0; r < 8; ++r) {
                    int m = mt * 16 + r + 8 * (lane >> 4);   // batch index
                    float v = c[r] + xproj_t[m * MLP_HID + n];
                    v = v > 0.0f ? v : 0.0f;
                    sH1[m * MLP_HID + n] = (__bf16)v;
                }
            } else if ((lane & 15) < 4) {                    // gate columns 128..131
                int g = lane & 15;
#pragma unroll
                for (int r = 0; r < 8; ++r) {
                    int m = mt * 16 + r + 8 * (lane >> 4);
                    sGpre[m * 4 + g] = c[r];
                }
            }
        }
        __syncthreads();

        // ---- Stage B: q = h1 @ W2 + b2  (256x32) via WMMA, 1 tile/wave ----
        {
            int mt = wave >> 1, ntq = wave & 1;
            v8f c = {};
#pragma unroll
            for (int kt = 0; kt < 4; ++kt) {
                v16bf a = load_a_tile(sH1 + mt * 16 * MLP_HID + kt * 32, MLP_HID, lane);
                v16bf b = load_b_frag(sW2F + (kt * 2 + ntq) * 512, lane);
                c = __builtin_amdgcn_wmma_f32_16x16x32_bf16(false, a, false, b,
                                                            (short)0, c, false, false);
            }
            int n = ntq * 16 + (lane & 15);
            float bias = sB2[n];
#pragma unroll
            for (int r = 0; r < 8; ++r) {
                int m = mt * 16 + r + 8 * (lane >> 4);
                sQ[m * 32 + n] = c[r] + bias;
            }
        }
        __syncthreads();

        // ---- Stage C: quantum gate (mean cumprod cos) + classical mix ----
        {
            int bb = tid >> 2, gg = tid & 3;
            float prod = 1.0f, acc = 0.0f;
#pragma unroll
            for (int j = 0; j < NQ; ++j) {
                prod *= __cosf(sQ[bb * 32 + gg * NQ + j]);
                acc  += prod;
            }
            float qv = acc * (1.0f / NQ);
            float cl = sGpre[bb * 4 + gg] + xg_t[bb * 4 + gg];
            float val = (gg == 2) ? (0.5f * tanhf(cl) + 0.5f * tanhf(qv))
                                  : (0.5f * sigmoidf_(cl) + 0.5f * sigmoidf_(qv));
            sGate[gg * BATCH + bb] = val;
        }
        __syncthreads();

        // ---- Stage D: cx/hx update + output write (64 elements/thread) ----
        float* out_t = out + (long)t * BATCH * HID_DIM;
        for (int e = tid; e < BATCH * HID_DIM; e += 1024) {
            int b = e >> 8;
            float fb = sGate[0 * BATCH + b];
            float ib = sGate[1 * BATCH + b];
            float gb = sGate[2 * BATCH + b];
            float ob = sGate[3 * BATCH + b];
            float cnew = fb * cxws[e] + ib * gb;
            cxws[e] = cnew;
            float hnew = ob * tanhf(cnew);
            out_t[e] = hnew;
            sHx[e] = (__bf16)hnew;                           // next-step A matrix
        }
        __syncthreads();
    }

    // ---- tail: final (hx, cx) after the outputs block ----
    float* hx_out = out + (long)SEQ * BATCH * HID_DIM;
    float* cx_out = hx_out + BATCH * HID_DIM;
    const float* last = out + (long)(SEQ - 1) * BATCH * HID_DIM;
    for (int e = tid; e < BATCH * HID_DIM; e += 1024) {
        hx_out[e] = last[e];
        cx_out[e] = cxws[e];
    }
}

// ---------------------------------------------------------------------------
extern "C" void kernel_launch(void* const* d_in, const int* in_sizes, int n_in,
                              void* d_out, int out_size, void* d_ws, size_t ws_size,
                              hipStream_t stream) {
    const float* x   = (const float*)d_in[0];
    const float* W1  = (const float*)d_in[1];
    const float* b1  = (const float*)d_in[2];
    const float* W2  = (const float*)d_in[3];
    const float* b2  = (const float*)d_in[4];
    const float* Wf  = (const float*)d_in[5];
    const float* bfv = (const float*)d_in[6];
    const float* Wi  = (const float*)d_in[7];
    const float* biv = (const float*)d_in[8];
    const float* Wg  = (const float*)d_in[9];
    const float* bgv = (const float*)d_in[10];
    const float* Wo  = (const float*)d_in[11];
    const float* bov = (const float*)d_in[12];
    float* out = (float*)d_out;

    char*  ws    = (char*)d_ws;
    float* xproj = (float*)ws;                                             // R*128 f32
    float* xg    = (float*)(ws + (size_t)RTOT * MLP_HID * sizeof(float));  // R*4 f32
    float* cxws  = (float*)(ws + (size_t)RTOT * MLP_HID * sizeof(float)
                               + (size_t)RTOT * 4 * sizeof(float));        // 256*256 f32

    // Phase 1: parallel x-projections incl. gate columns (one WMMA GEMM).
    size_t lds1 = (size_t)(8 * NT1 * 512 + 16 * IN_DIM) * sizeof(__bf16);
    qlstm_xproj_gemm<<<RTOT / 16, 256, lds1, stream>>>(
        x, W1, b1, Wf, bfv, Wi, biv, Wg, bgv, Wo, bov, xproj, xg);

    // Phase 2: persistent single-WGP recurrent kernel (32 waves).
    qlstm_recurrent<<<1, 1024, (size_t)LDS_TOTAL, stream>>>(
        xproj, xg, W1, W2, b2, Wf, Wi, Wg, Wo, cxws, out);
}
